// MultiHeadAttention_20701742366797
// MI455X (gfx1250) — matrix operand order repaired and checked by reading, not compiled, not run on hardware
//
#include <hip/hip_runtime.h>
#include <hip/hip_bf16.h>

#define DM   512
#define SLEN 2048
#define NH   8
#define DK   64

typedef __attribute__((ext_vector_type(16))) _Float16 v16h;
typedef __attribute__((ext_vector_type(8)))  _Float16 v8h;
typedef __attribute__((ext_vector_type(4)))  _Float16 v4h;
typedef __attribute__((ext_vector_type(8)))  float    v8f;

__device__ __forceinline__ v8f wmma_f16(v16h a, v16h b, v8f c) {
  // (neg_a, A, neg_b, B, c_mod, C, reuse_a, reuse_b)
  return __builtin_amdgcn_wmma_f32_16x16x32_f16(false, a, false, b, (short)0, c,
                                                false, false);
}

// A-matrix 16x32 f16 fragment (ISA 7.12.2): lane<16 holds K {kb+0..7, kb+16..23},
// lane>=16 holds K {kb+8..15, kb+24..31}; row = base + (lane&15).
__device__ __forceinline__ v16h load_a_frag(const _Float16* row, int kb, int aHi8) {
  v8h lo = *(const v8h*)(row + kb + aHi8);
  v8h hi = *(const v8h*)(row + kb + aHi8 + 16);
  v16h f;
#pragma unroll
  for (int i = 0; i < 8; ++i) { f[i] = lo[i]; f[i + 8] = hi[i]; }
  return f;
}

// B-matrix 32x16 f16 fragment: lane<16 holds K {kb+0..7, kb+16..23}, lane>=16 holds K {kb+8..15, kb+24..31},
// column n = (lane&15). "row" points at B[:,n] stored contiguously in k.
__device__ __forceinline__ v16h load_b_frag(const _Float16* row, int kb, int bHi16) {
  v8h lo = *(const v8h*)(row + kb + (bHi16 >> 1));
  v8h hi = *(const v8h*)(row + kb + (bHi16 >> 1) + 16);
  v16h f;
#pragma unroll
  for (int i = 0; i < 8; ++i) { f[i] = lo[i]; f[i + 8] = hi[i]; }
  return f;
}

// ---------------------------------------------------------------------------
// Projection GEMM: y = x @ W^T + b, fp32 in, f16 out in head-split [B,H,S,64].
// Block tile 128(M) x 64(N), 8 waves, K staged 64-deep per barrier (8 WMMA/wave).
// ---------------------------------------------------------------------------
__global__ __launch_bounds__(256) void proj_kernel(
    const float* __restrict__ x, const float* __restrict__ W,
    const float* __restrict__ bias, _Float16* __restrict__ out, float scale) {
  __shared__ _Float16 As[128 * 72];  // 64 k-halfs + 8 pad (144B rows, 16B aligned)
  __shared__ _Float16 Ws[64 * 72];

  const int tid    = threadIdx.x;
  const int lane   = tid & 31;
  const int wave   = tid >> 5;
  const int laneLo = lane & 15;
  const int aHi    = (lane >= 16) ? 8 : 0;
  const int bHi    = (lane >= 16) ? 16 : 0;
  const int m0     = blockIdx.x * 128;
  const int n0     = blockIdx.y * 64;

  v8f acc[4] = {};

  for (int kb = 0; kb < DM; kb += 64) {
    // Stage A: 128x64 fp32 -> f16. 2048 float4 slots, 8 per thread, coalesced.
#pragma unroll
    for (int i = 0; i < 8; ++i) {
      int slot = tid + i * 256;
      int row = slot >> 4, c4 = slot & 15;
      float4 v = *(const float4*)(x + (size_t)(m0 + row) * DM + kb + c4 * 4);
      v4h hh;
      hh[0] = (_Float16)v.x; hh[1] = (_Float16)v.y;
      hh[2] = (_Float16)v.z; hh[3] = (_Float16)v.w;
      *(v4h*)(As + row * 72 + c4 * 4) = hh;
    }
    // Stage W tile: rows n0..n0+63, cols kb..kb+63. 1024 float4 slots, 4/thread.
#pragma unroll
    for (int i = 0; i < 4; ++i) {
      int slot = tid + i * 256;
      int row = slot >> 4, c4 = slot & 15;
      float4 v = *(const float4*)(W + (size_t)(n0 + row) * DM + kb + c4 * 4);
      v4h hh;
      hh[0] = (_Float16)v.x; hh[1] = (_Float16)v.y;
      hh[2] = (_Float16)v.z; hh[3] = (_Float16)v.w;
      *(v4h*)(Ws + row * 72 + c4 * 4) = hh;
    }
    __syncthreads();

    const _Float16* arow = As + (wave * 16 + laneLo) * 72;
    v16h a0 = load_a_frag(arow, 0, aHi);
    v16h a1 = load_a_frag(arow, 32, aHi);
#pragma unroll
    for (int nt = 0; nt < 4; ++nt) {
      const _Float16* brow = Ws + (nt * 16 + laneLo) * 72;
      v16h b0 = load_b_frag(brow, 0, bHi);
      v16h b1 = load_b_frag(brow, 32, bHi);
      acc[nt] = wmma_f16(a0, b0, acc[nt]);
      acc[nt] = wmma_f16(a1, b1, acc[nt]);
    }
    __syncthreads();
  }

  // Epilogue: C layout -> f16 head-split store. Row = r (lanes<16) / r+8 (>=16).
  const int rAdd = (lane >= 16) ? 8 : 0;
#pragma unroll
  for (int nt = 0; nt < 4; ++nt) {
    const int n  = n0 + nt * 16 + laneLo;
    const float bn = bias[n];
    const int hh = n >> 6, d = n & 63;
#pragma unroll
    for (int r = 0; r < 8; ++r) {
      int m = m0 + wave * 16 + r + rAdd;
      int bb = m >> 11, s = m & (SLEN - 1);
      float val = (acc[nt][r] + bn) * scale;
      out[(((size_t)bb * NH + hh) * SLEN + s) * DK + d] = (_Float16)val;
    }
  }
}

// ---------------------------------------------------------------------------
// Flash attention: block = (b,h) x 64 query rows, 4 waves x 16 rows.
// Qh/Kh/Vh: f16 [B,H,S,64].  Out O: f16 [B,S,512] (heads merged).
// ---------------------------------------------------------------------------
__global__ __launch_bounds__(128) void attn_kernel(
    const _Float16* __restrict__ Qh, const _Float16* __restrict__ Kh,
    const _Float16* __restrict__ Vh, _Float16* __restrict__ O) {
  __shared__ _Float16 Vt[DK * 72];      // V transposed [d][key], pad 64->72
  __shared__ _Float16 Ps[4][16 * 72];   // per-wave P tile [row][key]

  const int tid    = threadIdx.x;
  const int lane   = tid & 31;
  const int wave   = tid >> 5;
  const int laneLo = lane & 15;
  const int aHi    = (lane >= 16) ? 8 : 0;
  const int bHi    = (lane >= 16) ? 16 : 0;
  const int rAdd   = (lane >= 16) ? 8 : 0;

  const int bh = blockIdx.y;            // b*8 + h
  const int b  = bh >> 3, h = bh & 7;
  const int q0 = blockIdx.x * 64 + wave * 16;

  const _Float16* Qs = Qh + (size_t)bh * SLEN * DK;
  const _Float16* Ks = Kh + (size_t)bh * SLEN * DK;
  const _Float16* Vs = Vh + (size_t)bh * SLEN * DK;

  // Q A-fragments (reused for the whole row of key blocks); [S,64] contiguous.
  v16h qf0 = load_a_frag(Qs + (size_t)(q0 + laneLo) * DK, 0, aHi);
  v16h qf1 = load_a_frag(Qs + (size_t)(q0 + laneLo) * DK, 32, aHi);

  v8f o[4] = {};
  float m_run[8], l_run[8];
#pragma unroll
  for (int r = 0; r < 8; ++r) { m_run[r] = -1e30f; l_run[r] = 0.0f; }

  for (int j = 0; j < SLEN; j += 64) {
    // Prefetch next key block (speculative; dropped if past the slice).
    if (j + 64 < SLEN) {
      __builtin_prefetch(Ks + (size_t)(j + 64 + (tid & 63)) * DK, 0, 1);
      __builtin_prefetch(Vs + (size_t)(j + 64 + (tid & 63)) * DK, 0, 1);
    }

    // Stage V block transposed: thread t -> key=t/2, 32 cols.
    {
      int key = tid >> 1;
      int c0  = (tid & 1) * 32;
      const v8h* vrow = (const v8h*)(Vs + (size_t)(j + key) * DK + c0);
#pragma unroll
      for (int cc = 0; cc < 4; ++cc) {
        v8h vv = vrow[cc];
#pragma unroll
        for (int e = 0; e < 8; ++e) Vt[(c0 + cc * 8 + e) * 72 + key] = vv[e];
      }
    }

    // Scores S = Q' K^T : 4 key-subtiles x (K=64 -> 2 wmma). B-frags from global.
    v8f sc[4];
#pragma unroll
    for (int nt = 0; nt < 4; ++nt) {
      const _Float16* krow = Ks + (size_t)(j + nt * 16 + laneLo) * DK;
      v16h bk0 = load_b_frag(krow, 0, bHi);
      v16h bk1 = load_b_frag(krow, 32, bHi);
      v8f s = {};
      s = wmma_f16(qf0, bk0, s);
      s = wmma_f16(qf1, bk1, s);
      sc[nt] = s;
    }

    // Online softmax. Row r lives in lanes 0-15 (M=r) / 16-31 (M=r+8);
    // xor masks 1,2,4,8 reduce over the 16 columns without crossing halves.
#pragma unroll
    for (int r = 0; r < 8; ++r) {
      float tmax = fmaxf(fmaxf(sc[0][r], sc[1][r]), fmaxf(sc[2][r], sc[3][r]));
#pragma unroll
      for (int mask = 1; mask < 16; mask <<= 1)
        tmax = fmaxf(tmax, __shfl_xor(tmax, mask, 32));
      float m_new = fmaxf(m_run[r], tmax);
      float corr  = __expf(m_run[r] - m_new);
      m_run[r] = m_new;
      float rsum = 0.0f;
#pragma unroll
      for (int nt = 0; nt < 4; ++nt) {
        float p = __expf(sc[nt][r] - m_new);
        rsum += p;
        Ps[wave][(r + rAdd) * 72 + nt * 16 + laneLo] = (_Float16)p;
        o[nt][r] *= corr;
      }
#pragma unroll
      for (int mask = 1; mask < 16; mask <<= 1)
        rsum += __shfl_xor(rsum, mask, 32);
      l_run[r] = l_run[r] * corr + rsum;
    }
    __syncthreads();  // Vt + Ps visible

    // O += P @ V : P as A (K=keys), Vt rows give B columns contiguously.
    const _Float16* prow = &Ps[wave][laneLo * 72];
    v16h pf0 = load_a_frag(prow, 0, aHi);
    v16h pf1 = load_a_frag(prow, 32, aHi);
#pragma unroll
    for (int nt = 0; nt < 4; ++nt) {
      const _Float16* vtrow = Vt + (nt * 16 + laneLo) * 72;
      v16h bv0 = load_b_frag(vtrow, 0, bHi);
      v16h bv1 = load_b_frag(vtrow, 32, bHi);
      o[nt] = wmma_f16(pf0, bv0, o[nt]);
      o[nt] = wmma_f16(pf1, bv1, o[nt]);
    }
    __syncthreads();  // safe to restage Vt next iteration
  }

  // Normalize and store f16 [B,S,512] (heads merged).
#pragma unroll
  for (int nt = 0; nt < 4; ++nt) {
    const int d = nt * 16 + laneLo;
#pragma unroll
    for (int r = 0; r < 8; ++r) {
      int srow = q0 + r + rAdd;
      float val = o[nt][r] / l_run[r];
      O[((size_t)b * SLEN + srow) * DM + h * DK + d] = (_Float16)val;
    }
  }
}

// ---------------------------------------------------------------------------
// Output projection: out = O @ Wo^T + bo, f16 activations in, fp32 out.
// A tile staged with CDNA5 async global->LDS copies (ASYNCcnt-tracked).
// ---------------------------------------------------------------------------
__global__ __launch_bounds__(256) void outproj_kernel(
    const _Float16* __restrict__ xO, const float* __restrict__ W,
    const float* __restrict__ bias, float* __restrict__ out) {
  __shared__ _Float16 As[128 * 72];
  __shared__ _Float16 Ws[64 * 72];

  const int tid    = threadIdx.x;
  const int lane   = tid & 31;
  const int wave   = tid >> 5;
  const int laneLo = lane & 15;
  const int aHi    = (lane >= 16) ? 8 : 0;
  const int bHi    = (lane >= 16) ? 16 : 0;
  const int m0     = blockIdx.x * 128;
  const int n0     = blockIdx.y * 64;

  v8f acc[4] = {};

  for (int kb = 0; kb < DM; kb += 64) {
    // Stage A: 128x64 halfs = 1024 16B chunks, 4 per thread, async to LDS.
#pragma unroll
    for (int i = 0; i < 4; ++i) {
      int slot = tid + i * 256;
      int row = slot >> 3, c8 = slot & 7;
      unsigned lds = (unsigned)(uintptr_t)(As + row * 72 + c8 * 8);
      unsigned long long ga =
          (unsigned long long)(uintptr_t)(xO + (size_t)(m0 + row) * DM + kb + c8 * 8);
      asm volatile("global_load_async_to_lds_b128 %0, %1, off"
                   :: "v"(lds), "v"(ga) : "memory");
    }
    // Stage W tile (needs f32->f16 conversion, regular path).
#pragma unroll
    for (int i = 0; i < 4; ++i) {
      int slot = tid + i * 256;
      int row = slot >> 4, c4 = slot & 15;
      float4 v = *(const float4*)(W + (size_t)(n0 + row) * DM + kb + c4 * 4);
      v4h hh;
      hh[0] = (_Float16)v.x; hh[1] = (_Float16)v.y;
      hh[2] = (_Float16)v.z; hh[3] = (_Float16)v.w;
      *(v4h*)(Ws + row * 72 + c4 * 4) = hh;
    }
    asm volatile("s_wait_asynccnt 0" ::: "memory");
    __syncthreads();

    const _Float16* arow = As + (wave * 16 + laneLo) * 72;
    v16h a0 = load_a_frag(arow, 0, aHi);
    v16h a1 = load_a_frag(arow, 32, aHi);
#pragma unroll
    for (int nt = 0; nt < 4; ++nt) {
      const _Float16* brow = Ws + (nt * 16 + laneLo) * 72;
      v16h b0 = load_b_frag(brow, 0, bHi);
      v16h b1 = load_b_frag(brow, 32, bHi);
      acc[nt] = wmma_f16(a0, b0, acc[nt]);
      acc[nt] = wmma_f16(a1, b1, acc[nt]);
    }
    __syncthreads();
  }

  const int rAdd = (lane >= 16) ? 8 : 0;
#pragma unroll
  for (int nt = 0; nt < 4; ++nt) {
    const int n = n0 + nt * 16 + laneLo;
    const float bn = bias[n];
#pragma unroll
    for (int r = 0; r < 8; ++r) {
      int m = m0 + wave * 16 + r + rAdd;
      out[(size_t)m * DM + n] = acc[nt][r] + bn;
    }
  }
}

extern "C" void kernel_launch(void* const* d_in, const int* in_sizes, int n_in,
                              void* d_out, int out_size, void* d_ws, size_t ws_size,
                              hipStream_t stream) {
  (void)in_sizes; (void)n_in; (void)out_size; (void)ws_size;

  const float* q  = (const float*)d_in[0];
  const float* k  = (const float*)d_in[1];
  const float* v  = (const float*)d_in[2];
  const float* Wq = (const float*)d_in[3];
  const float* bq = (const float*)d_in[4];
  const float* Wk = (const float*)d_in[5];
  const float* bk = (const float*)d_in[6];
  const float* Wv = (const float*)d_in[7];
  const float* bv = (const float*)d_in[8];
  const float* Wo = (const float*)d_in[9];
  const float* bo = (const float*)d_in[10];
  float* out = (float*)d_out;

  const size_t perTensor = (size_t)4 * NH * SLEN * DK;  // 4,194,304 f16 elems
  _Float16* Qh = (_Float16*)d_ws;
  _Float16* Kh = Qh + perTensor;
  _Float16* Vh = Kh + perTensor;
  _Float16* Oc = Vh + perTensor;  // f16 [B,S,512]

  dim3 gGemm(64, 8), bGemm(256);
  // Fold 1/sqrt(d_k)=0.125 into the Q projection.
  proj_kernel<<<gGemm, bGemm, 0, stream>>>(q, Wq, bq, Qh, 0.125f);
  proj_kernel<<<gGemm, bGemm, 0, stream>>>(k, Wk, bk, Kh, 1.0f);
  proj_kernel<<<gGemm, bGemm, 0, stream>>>(v, Wv, bv, Vh, 1.0f);

  attn_kernel<<<dim3(SLEN / 64, 4 * NH), 128, 0, stream>>>(Qh, Kh, Vh, Oc);

  outproj_kernel<<<gGemm, bGemm, 0, stream>>>(Oc, Wo, bo, out);
}
